// LearnedGradientDescent_944892805807
// MI455X (gfx1250) — matrix operand order repaired and checked by reading, not compiled, not run on hardware
//
#include <hip/hip_runtime.h>
#include <stdint.h>

#define BATCH 4
#define HH 256
#define WW 256
#define NPIX 65536
#define NMEM 5
#define NITER 10
#define PI_F 3.14159265358979323846f

typedef __attribute__((ext_vector_type(16))) _Float16 v16h;
typedef __attribute__((ext_vector_type(8)))  float    v8f;

#if __has_builtin(__builtin_amdgcn_tensor_load_to_lds) && __has_builtin(__builtin_amdgcn_s_wait_tensorcnt)
#define LGD_HAVE_TDM 1
typedef __attribute__((ext_vector_type(4))) unsigned lgd_u32x4;
typedef __attribute__((ext_vector_type(8))) int      lgd_i32x8;
typedef __attribute__((ext_vector_type(4))) int      lgd_i32x4;
#else
#define LGD_HAVE_TDM 0
#endif

__device__ __forceinline__ float2 cmulf(float2 a, float2 b) {
    return make_float2(a.x * b.x - a.y * b.y, a.x * b.y + a.y * b.x);
}

// ---------------------------------------------------------------------------
// 256-point Stockham radix-2 FFT in LDS. 128 threads per FFT, block-wide syncs
// (both sub-FFTs of a block run the same loop). Result ends in X.
// ---------------------------------------------------------------------------
__device__ float2* fft256_stages(float2* X, float2* Y, int t) {
    float2 *px = X, *py = Y;
#pragma unroll
    for (int st = 0; st < 8; ++st) {
        int s = 1 << st;
        int m = 128 >> st;
        int q = t & (s - 1);
        int p = t >> st;
        float2 a = px[q + s * p];
        float2 c = px[q + s * (p + m)];
        float ang = -PI_F * (float)p / (float)m; // exp(-2*pi*i*p/(2m))
        float sn, cs;
        __sincosf(ang, &sn, &cs);
        float2 w = make_float2(cs, sn);
        py[q + s * 2 * p]       = make_float2(a.x + c.x, a.y + c.y);
        py[q + s * (2 * p + 1)] = cmulf(make_float2(a.x - c.x, a.y - c.y), w);
        float2* tm = px; px = py; py = tm;
        __syncthreads();
    }
    return px;
}

// Column FFT over n1 (stride 256). gather=1: real input coeff[b, order[n]].
__global__ __launch_bounds__(256) void lgd_fft_col(
    const float* __restrict__ coeff, const int* __restrict__ order,
    const float2* __restrict__ cin, float2* __restrict__ cout, int gather) {
    __shared__ float2 lds[2][2][256];
    int sub = threadIdx.x >> 7;
    int t = threadIdx.x & 127;
    int colg = blockIdx.x * 2 + sub;     // among BATCH*256
    int b = colg >> 8;
    int n2 = colg & 255;
    float2* X = lds[sub][0];
    float2* Y = lds[sub][1];
    for (int i = t; i < 256; i += 128) {
        int n = i * 256 + n2;
        float2 v;
        if (gather) { v.x = coeff[b * NPIX + order[n]]; v.y = 0.0f; }
        else        { v = cin[b * NPIX + n]; }
        X[i] = v;
    }
    __syncthreads();
    float2* res = fft256_stages(X, Y, t);
    for (int i = t; i < 256; i += 128)
        cout[b * NPIX + i * 256 + n2] = res[i];
}

// Row FFT over n2 (contiguous) with four-step twiddle exp(-2*pi*i*k1*n2/65536).
__global__ __launch_bounds__(256) void lgd_fft_row(
    const float2* __restrict__ cin, float2* __restrict__ cout) {
    __shared__ float2 lds[2][2][256];
    int sub = threadIdx.x >> 7;
    int t = threadIdx.x & 127;
    int rowg = blockIdx.x * 2 + sub;
    int b = rowg >> 8;
    int k1 = rowg & 255;
    float2* X = lds[sub][0];
    float2* Y = lds[sub][1];
    for (int i = t; i < 256; i += 128) {
        float2 v = cin[b * NPIX + k1 * 256 + i];
        float ang = -(2.0f * PI_F) * (float)(k1 * i) * (1.0f / 65536.0f);
        float sn, cs;
        __sincosf(ang, &sn, &cs);
        X[i] = cmulf(v, make_float2(cs, sn));
    }
    __syncthreads();
    float2* res = fft256_stages(X, Y, t);
    for (int i = t; i < 256; i += 128)
        cout[b * NPIX + k1 * 256 + i] = res[i];
}

// S[n] = conj(X[n]*Phi[n] - F_obs[n]) * Phi[n];  X read transposed from Bmat.
__global__ __launch_bounds__(256) void lgd_pointwise_mid(
    const float2* __restrict__ bm, const float* __restrict__ fr,
    const float* __restrict__ fi, const float* __restrict__ pr,
    const float* __restrict__ pi, float2* __restrict__ smat) {
    int i = blockIdx.x * 256 + threadIdx.x;
    if (i >= BATCH * NPIX) return;
    int b = i >> 16, n = i & 65535;
    float2 X = bm[b * NPIX + (n & 255) * 256 + (n >> 8)];
    float2 Phi = make_float2(pr[n], pi[n]);
    float2 R = cmulf(X, Phi);
    R.x -= fr[b * NPIX + n];
    R.y -= fi[b * NPIX + n];
    float2 Rc = make_float2(R.x, -R.y);
    smat[b * NPIX + n] = cmulf(Rc, Phi);
}

// grad[b, order[n]] = 2 * Re(C[n]) with C read transposed.
__global__ __launch_bounds__(256) void lgd_scatter_grad(
    const float2* __restrict__ cm, const int* __restrict__ order,
    float* __restrict__ grad) {
    int i = blockIdx.x * 256 + threadIdx.x;
    if (i >= BATCH * NPIX) return;
    int b = i >> 16, n = i & 65535;
    float2 G = cm[b * NPIX + (n & 255) * 256 + (n >> 8)];
    grad[b * NPIX + order[n]] = 2.0f * G.x;
}

// cnn_in (8ch f32): [coeff, data_grad, 2*lambda*coeff, memory(5)]
__global__ __launch_bounds__(256) void lgd_build_cnn_in(
    const float* __restrict__ coeff, const float* __restrict__ grad,
    const float* __restrict__ mem, const float* __restrict__ lam,
    float* __restrict__ cin) {
    int i = blockIdx.x * 256 + threadIdx.x;
    if (i >= BATCH * NPIX) return;
    int b = i >> 16, p = i & 65535;
    float c = coeff[i];
    cin[((size_t)(b * 8 + 0)) * NPIX + p] = c;
    cin[((size_t)(b * 8 + 1)) * NPIX + p] = grad[i];
    cin[((size_t)(b * 8 + 2)) * NPIX + p] = 2.0f * c * lam[0];
#pragma unroll
    for (int m = 0; m < NMEM; ++m)
        cin[((size_t)(b * 8 + 3 + m)) * NPIX + p] = mem[((size_t)(b * NMEM + m)) * NPIX + p];
}

// Per-(b,c) mean / rstd over 65536 pixels. gridDim.x = B*C.
__global__ __launch_bounds__(256) void lgd_stats(
    const float* __restrict__ x, float* __restrict__ stats) {
    int bc = blockIdx.x;
    const float* xp = x + (size_t)bc * NPIX;
    float s = 0.0f, s2 = 0.0f;
    for (int i = threadIdx.x; i < NPIX; i += 256) {
        float v = xp[i];
        s += v; s2 += v * v;
    }
    __shared__ float sh0[256], sh1[256];
    sh0[threadIdx.x] = s; sh1[threadIdx.x] = s2;
    __syncthreads();
    for (int o = 128; o > 0; o >>= 1) {
        if (threadIdx.x < o) {
            sh0[threadIdx.x] += sh0[threadIdx.x + o];
            sh1[threadIdx.x] += sh1[threadIdx.x + o];
        }
        __syncthreads();
    }
    if (threadIdx.x == 0) {
        float mean = sh0[0] * (1.0f / (float)NPIX);
        float var = sh1[0] * (1.0f / (float)NPIX) - mean * mean;
        stats[bc * 2 + 0] = mean;
        stats[bc * 2 + 1] = rsqrtf(var + 1e-5f);
    }
}

// InstanceNorm (+optional ReLU) and quantize to f16 conv input.
__global__ __launch_bounds__(256) void lgd_normalize(
    const float* __restrict__ x, const float* __restrict__ stats,
    const float* __restrict__ g, const float* __restrict__ bta,
    _Float16* __restrict__ out, int C, int relu) {
    size_t i = (size_t)blockIdx.x * 256 + threadIdx.x;
    if (i >= (size_t)BATCH * C * NPIX) return;
    int bc = (int)(i >> 16);
    int c = bc % C;
    float mean = stats[bc * 2 + 0];
    float rstd = stats[bc * 2 + 1];
    float v = (x[i] - mean) * rstd * g[c] + bta[c];
    if (relu) v = fmaxf(v, 0.0f);
    out[i] = (_Float16)v;
}

// Weight f32 -> f16, layout [tap(9)][ci_pad][co_pad], zero padded.
__global__ __launch_bounds__(256) void lgd_wcvt(
    const float* __restrict__ w, _Float16* __restrict__ wout,
    int Cout, int Cin, int CoutPad, int CinPad) {
    int total = 9 * CinPad * CoutPad;
    int i = blockIdx.x * 256 + threadIdx.x;
    if (i >= total) return;
    int co = i % CoutPad;
    int ci = (i / CoutPad) % CinPad;
    int tap = i / (CoutPad * CinPad);
    float v = 0.0f;
    if (co < Cout && ci < Cin) v = w[(co * Cin + ci) * 9 + tap];
    wout[i] = (_Float16)v;
}

__global__ __launch_bounds__(256) void lgd_init_state(
    const float* __restrict__ coeff0, float* __restrict__ coeff,
    float* __restrict__ mem, float* __restrict__ hist0) {
    int i = blockIdx.x * 256 + threadIdx.x;
    if (i < BATCH * NMEM * NPIX) mem[i] = 0.0f;
    if (i < BATCH * NPIX) {
        float v = coeff0[i];
        coeff[i] = v;
        hist0[i] = v;
    }
}

__global__ __launch_bounds__(256) void lgd_final_update(
    const float* __restrict__ c5, float* __restrict__ coeff,
    float* __restrict__ mem, const float* __restrict__ step,
    float* __restrict__ hist_slot, float* __restrict__ out_final) {
    int i = blockIdx.x * 256 + threadIdx.x;
    if (i >= BATCH * NPIX) return;
    int b = i >> 16, p = i & 65535;
    float nc = coeff[i] - c5[((size_t)(b * 6 + 0)) * NPIX + p] * step[0];
    coeff[i] = nc;
    hist_slot[i] = nc;
    out_final[i] = nc;
#pragma unroll
    for (int m = 0; m < NMEM; ++m)
        mem[((size_t)(b * NMEM + m)) * NPIX + p] =
            fmaxf(c5[((size_t)(b * 6 + 1 + m)) * NPIX + p], 0.0f);
}

// ---------------------------------------------------------------------------
// Implicit-GEMM dilated 3x3 conv with V_WMMA_F32_16X16X32_F16.
// Block = 256 threads (8 waves). Output tile: (MT*16) co x 128 pixels of one
// image row. Wave w owns pixel sub-tile w (16 px) and loops over MT m-tiles.
// K loop: 9 taps x (CINPAD/32) ci-chunks.
// A (weight) tile is DMA'd into LDS by the Tensor Data Mover (TDM): a
// contiguous 1-row [32ci x CoutPad] f16 tile, issued by wave 0, tracked with
// TENSORcnt. B (image) tile stays on predicated VMEM loads (zero padding).
// ---------------------------------------------------------------------------
template <int MT, int COUT, int CIN, int CINPAD, int DIL>
__global__ __launch_bounds__(256) void lgd_conv_wmma(
    const _Float16* __restrict__ in, const _Float16* __restrict__ wgt,
    const float* __restrict__ bias, float* __restrict__ out) {
    const int tile = blockIdx.x;
    const int x0 = (tile & 1) * 128;
    const int y  = (tile >> 1) & 255;
    const int b  = tile >> 9;
    constexpr int CoutPad = MT * 16;
    constexpr int AElems = 32 * CoutPad;     // f16 elems per A tile

    __shared__ __attribute__((aligned(16))) _Float16 Ald[AElems];
    __shared__ __attribute__((aligned(16))) _Float16 Bld[32 * 128]; // k-pair interleaved

    const int t = threadIdx.x;
    const int lane = t & 31;
    const int wv = t >> 5;
    const int half16 = lane >> 4;
    const int l16 = lane & 15;

    v8f acc[MT];
    const v8f zero = {0.f, 0.f, 0.f, 0.f, 0.f, 0.f, 0.f, 0.f};
#pragma unroll
    for (int m = 0; m < MT; ++m) acc[m] = zero;

    constexpr int NCHUNK = CINPAD >> 5;
    for (int tap = 0; tap < 9; ++tap) {
        const int ky = tap / 3 - 1, kx = tap % 3 - 1;
        const int yy = y + DIL * ky;
        const int xb = x0 + DIL * kx;
        const bool yok = (yy >= 0) && (yy < HH);
#pragma unroll
        for (int ch = 0; ch < NCHUNK; ++ch) {
            const int ci0 = ch << 5;
#if LGD_HAVE_TDM
            // --- A tile via Tensor Data Mover (wave 0 only) ---
            if (wv == 0) {
                unsigned long long ga =
                    (unsigned long long)(uintptr_t)wgt +
                    (unsigned long long)((tap * CINPAD + ci0) * CoutPad) * 2ull;
                unsigned ldsa = (unsigned)(uintptr_t)(void*)Ald;
                // D# group0: count=1 | lds_addr | global_addr(57b) | type=2
                lgd_u32x4 g0 = {
                    1u,
                    ldsa,
                    (unsigned)(ga & 0xffffffffu),
                    (unsigned)((ga >> 32) & 0x1ffffffu) | (2u << 30)
                };
                // D# group1: data_size=2B; tensor_dim0=tile_dim0=AElems;
                // tensor_dim1=tile_dim1=1; stride0=AElems.
                lgd_i32x8 g1 = {
                    (int)(1u << 16),                       // data_size=1 (2 bytes)
                    (int)(((unsigned)AElems & 0xffffu) << 16), // tensor_dim0 lo16
                    (int)(((unsigned)AElems >> 16) | (1u << 16)), // dim0 hi | dim1 lo=1
                    (int)(((unsigned)AElems & 0xffffu) << 16),    // dim1 hi=0 | tile_dim0
                    1,                                     // tile_dim1=1, tile_dim2=0
                    (int)AElems,                           // tensor_dim0_stride lo32
                    0, 0
                };
                lgd_i32x4 gz4 = {0, 0, 0, 0};
                lgd_i32x8 gz8 = {0, 0, 0, 0, 0, 0, 0, 0};
                __builtin_amdgcn_tensor_load_to_lds(g0, g1, gz4, gz4, gz8, 0);
            }
#else
            // --- A tile: manual staging [co][ci], wgt[tap][CINPAD][CoutPad] ---
            for (int i = t; i < AElems; i += 256) {
                int ci = i & 31, co = i >> 5;
                Ald[co * 32 + ci] = wgt[(tap * CINPAD + ci0 + ci) * CoutPad + co];
            }
#endif
            // --- B tile: 32 ci x 128 pixels, k-pair interleaved for 32b reads ---
            for (int i = t; i < 32 * 128; i += 256) {
                int p = i & 127, ci = i >> 7;
                int xx = xb + p;
                _Float16 v = (_Float16)0.0f;
                if (yok && xx >= 0 && xx < WW && (ci0 + ci) < CIN)
                    v = in[(((size_t)(b * CIN + ci0 + ci)) * HH + yy) * WW + xx];
                Bld[((ci >> 1) * 128 + p) * 2 + (ci & 1)] = v;
            }
#if LGD_HAVE_TDM
            if (wv == 0) __builtin_amdgcn_s_wait_tensorcnt(0);
#endif
            __syncthreads();
            // B fragment: column = pixel; lanes 0-15 K 0..7,16..23, lanes 16-31 K 8..15,24..31
            union { v16h v; unsigned u[8]; } bf;
            {
                int p = wv * 16 + l16;
                int kb2 = half16 * 4;
                const unsigned* Bu = (const unsigned*)Bld;
#pragma unroll
                for (int j = 0; j < 8; ++j) bf.u[j] = Bu[(((j < 4) ? 0 : 8) + kb2 + (j & 3)) * 128 + p];
            }
#pragma unroll
            for (int m = 0; m < MT; ++m) {
                union { v16h v; unsigned u[8]; } af;
                int row = m * 16 + l16;
                int kbase = half16 * 8;
                const unsigned* Au = (const unsigned*)Ald;
#pragma unroll
                for (int j = 0; j < 8; ++j) {
                    int k0 = ((j < 4) ? 0 : 16) + kbase + 2 * (j & 3);
                    af.u[j] = Au[(row * 32 + k0) >> 1];
                }
                acc[m] = __builtin_amdgcn_wmma_f32_16x16x32_f16(
                    false, af.v, false, bf.v, (short)0, acc[m], false, false);
            }
            __syncthreads();
        }
    }
    // Store D: lane (l16)=pixel col, VGPR r -> co = m*16 + half16*8 + r
    const int x = x0 + wv * 16 + l16;
#pragma unroll
    for (int m = 0; m < MT; ++m) {
#pragma unroll
        for (int r = 0; r < 8; ++r) {
            int co = m * 16 + half16 * 8 + r;
            if (co < COUT)
                out[(((size_t)(b * COUT + co)) * HH + y) * WW + x] =
                    acc[m][r] + bias[co];
        }
    }
}

// ---------------------------------------------------------------------------
extern "C" void kernel_launch(void* const* d_in, const int* in_sizes, int n_in,
                              void* d_out, int out_size, void* d_ws, size_t ws_size,
                              hipStream_t stream) {
    (void)in_sizes; (void)n_in; (void)out_size; (void)ws_size;

    const float* coeff0 = (const float*)d_in[0];
    const float* fobs_r = (const float*)d_in[1];
    const float* fobs_i = (const float*)d_in[2];
    const float* phi_r  = (const float*)d_in[3];
    const float* phi_i  = (const float*)d_in[4];
    const int*   order  = (const int*)d_in[5];
    const float* in0_g = (const float*)d_in[6];
    const float* in0_b = (const float*)d_in[7];
    const float* w1 = (const float*)d_in[8];
    const float* b1 = (const float*)d_in[9];
    const float* in1_g = (const float*)d_in[10];
    const float* in1_b = (const float*)d_in[11];
    const float* w2 = (const float*)d_in[12];
    const float* b2 = (const float*)d_in[13];
    const float* in2_g = (const float*)d_in[14];
    const float* in2_b = (const float*)d_in[15];
    const float* w3 = (const float*)d_in[16];
    const float* b3 = (const float*)d_in[17];
    const float* in3_g = (const float*)d_in[18];
    const float* in3_b = (const float*)d_in[19];
    const float* w4 = (const float*)d_in[20];
    const float* b4 = (const float*)d_in[21];
    const float* in4_g = (const float*)d_in[22];
    const float* in4_b = (const float*)d_in[23];
    const float* w5 = (const float*)d_in[24];
    const float* b5 = (const float*)d_in[25];
    const float* lam  = (const float*)d_in[26];
    const float* step = (const float*)d_in[27];

    char* base = (char*)d_ws;
    size_t off = 0;
    auto alloc = [&](size_t bytes) {
        char* p = base + off;
        off = (off + bytes + 255) & ~(size_t)255;
        return p;
    };
    float*     coeff_w = (float*)alloc((size_t)BATCH * NPIX * 4);
    float*     memory  = (float*)alloc((size_t)BATCH * NMEM * NPIX * 4);
    float*     grad    = (float*)alloc((size_t)BATCH * NPIX * 4);
    float*     cnn_in  = (float*)alloc((size_t)BATCH * 8 * NPIX * 4);
    _Float16*  bufIn   = (_Float16*)alloc((size_t)BATCH * 64 * NPIX * 2);
    float*     bufOut  = (float*)alloc((size_t)BATCH * 64 * NPIX * 4);
    float2*    c0      = (float2*)alloc((size_t)BATCH * NPIX * 8);
    float2*    c1      = (float2*)alloc((size_t)BATCH * NPIX * 8);
    _Float16*  wgt1    = (_Float16*)alloc(9 * 32 * 64 * 2);
    _Float16*  wgt2    = (_Float16*)alloc(9 * 64 * 64 * 2);
    _Float16*  wgt3    = (_Float16*)alloc(9 * 64 * 64 * 2);
    _Float16*  wgt4    = (_Float16*)alloc(9 * 64 * 64 * 2);
    _Float16*  wgt5    = (_Float16*)alloc(9 * 64 * 16 * 2);
    float*     stats   = (float*)alloc((size_t)BATCH * 64 * 2 * 4);

    float* out_coeff = (float*)d_out;
    float* hist = out_coeff + (size_t)BATCH * NPIX;

    const dim3 blk(256);
    const int gBN   = BATCH * NPIX / 256;             // 1024
    const int gFFT  = BATCH * 256 / 2;                // 512
    const int gConv = BATCH * HH * (WW / 128);        // 2048

    lgd_init_state<<<BATCH * NMEM * NPIX / 256, blk, 0, stream>>>(coeff0, coeff_w, memory, hist);
    lgd_wcvt<<<(9 * 32 * 64 + 255) / 256, blk, 0, stream>>>(w1, wgt1, 64, 8, 64, 32);
    lgd_wcvt<<<(9 * 64 * 64 + 255) / 256, blk, 0, stream>>>(w2, wgt2, 64, 64, 64, 64);
    lgd_wcvt<<<(9 * 64 * 64 + 255) / 256, blk, 0, stream>>>(w3, wgt3, 64, 64, 64, 64);
    lgd_wcvt<<<(9 * 64 * 64 + 255) / 256, blk, 0, stream>>>(w4, wgt4, 64, 64, 64, 64);
    lgd_wcvt<<<(9 * 64 * 16 + 255) / 256, blk, 0, stream>>>(w5, wgt5, 6, 64, 16, 64);

    for (int it = 0; it < NITER; ++it) {
        // ---- data fidelity gradient via four-step FFTs ----
        lgd_fft_col<<<gFFT, blk, 0, stream>>>(coeff_w, order, nullptr, c0, 1);
        lgd_fft_row<<<gFFT, blk, 0, stream>>>(c0, c1);
        lgd_pointwise_mid<<<gBN, blk, 0, stream>>>(c1, fobs_r, fobs_i, phi_r, phi_i, c0);
        lgd_fft_col<<<gFFT, blk, 0, stream>>>(nullptr, order, c0, c1, 0);
        lgd_fft_row<<<gFFT, blk, 0, stream>>>(c1, c0);
        lgd_scatter_grad<<<gBN, blk, 0, stream>>>(c0, order, grad);

        // ---- CNN ----
        lgd_build_cnn_in<<<gBN, blk, 0, stream>>>(coeff_w, grad, memory, lam, cnn_in);
        lgd_stats<<<BATCH * 8, blk, 0, stream>>>(cnn_in, stats);
        lgd_normalize<<<BATCH * 8 * NPIX / 256, blk, 0, stream>>>(cnn_in, stats, in0_g, in0_b, bufIn, 8, 0);
        lgd_conv_wmma<4, 64, 8, 32, 1><<<gConv, blk, 0, stream>>>(bufIn, wgt1, b1, bufOut);

        lgd_stats<<<BATCH * 64, blk, 0, stream>>>(bufOut, stats);
        lgd_normalize<<<BATCH * 64 * NPIX / 256, blk, 0, stream>>>(bufOut, stats, in1_g, in1_b, bufIn, 64, 1);
        lgd_conv_wmma<4, 64, 64, 64, 2><<<gConv, blk, 0, stream>>>(bufIn, wgt2, b2, bufOut);

        lgd_stats<<<BATCH * 64, blk, 0, stream>>>(bufOut, stats);
        lgd_normalize<<<BATCH * 64 * NPIX / 256, blk, 0, stream>>>(bufOut, stats, in2_g, in2_b, bufIn, 64, 1);
        lgd_conv_wmma<4, 64, 64, 64, 4><<<gConv, blk, 0, stream>>>(bufIn, wgt3, b3, bufOut);

        lgd_stats<<<BATCH * 64, blk, 0, stream>>>(bufOut, stats);
        lgd_normalize<<<BATCH * 64 * NPIX / 256, blk, 0, stream>>>(bufOut, stats, in3_g, in3_b, bufIn, 64, 1);
        lgd_conv_wmma<4, 64, 64, 64, 8><<<gConv, blk, 0, stream>>>(bufIn, wgt4, b4, bufOut);

        lgd_stats<<<BATCH * 64, blk, 0, stream>>>(bufOut, stats);
        lgd_normalize<<<BATCH * 64 * NPIX / 256, blk, 0, stream>>>(bufOut, stats, in4_g, in4_b, bufIn, 64, 1);
        lgd_conv_wmma<1, 6, 64, 64, 1><<<gConv, blk, 0, stream>>>(bufIn, wgt5, b5, bufOut);

        lgd_final_update<<<gBN, blk, 0, stream>>>(
            bufOut, coeff_w, memory, step,
            hist + (size_t)(it + 1) * BATCH * NPIX, out_coeff);
    }
}
